// MultiScaleModeling_35605278884498
// MI455X (gfx1250) — matrix operand order repaired and checked by reading, not compiled, not run on hardware
//
#include <hip/hip_runtime.h>
#include <cstdint>
#include <cstddef>

#define B_  2
#define DZ_ 41
#define DY_ 400
#define DX_ 400
#define N_  200000
#define GRID_TOTAL (B_*DZ_*DY_*DX_)

typedef __attribute__((ext_vector_type(16))) __bf16 v16bf;
typedef __attribute__((ext_vector_type(8)))  float  v8f;

// Packed kernel offsets for all 88 taps: (oz+1) | (oy+3)<<2 | (ox+3)<<5,
// in reference iteration order (iz outer, iy, ix inner) per group.
// g0: 1x3x3 dil1 (9) | g1: 1x5x5 dil1 (25) | g2: 3x3x3 dil1 (27) | g3: 3x3x3 dil(1,3,3) (27)
__device__ const int OFF88[88] = {
  // g0 (gob=0)
  73,105,137, 77,109,141, 81,113,145,
  // g1 (gob=9)
  37,69,101,133,165, 41,73,105,137,169, 45,77,109,141,173,
  49,81,113,145,177, 53,85,117,149,181,
  // g2 (gob=34)
  72,104,136, 76,108,140, 80,112,144,
  73,105,137, 77,109,141, 81,113,145,
  74,106,138, 78,110,142, 82,114,146,
  // g3 (gob=61)
  0,96,192, 12,108,204, 24,120,216,
  1,97,193, 13,109,205, 25,121,217,
  2,98,194, 14,110,206, 26,122,218
};

__global__ void fill_grid(int* __restrict__ g) {
  int i = blockIdx.x * blockDim.x + threadIdx.x;
  if (i < GRID_TOTAL) g[i] = -1;
}

__global__ void scatter_grid(const int* __restrict__ idx, int* __restrict__ g) {
  int i = blockIdx.x * blockDim.x + threadIdx.x;
  if (i < N_) {
    int4 c = *(const int4*)(idx + i * 4);              // (b,z,y,x)
    g[((c.x * DZ_ + c.y) * DY_ + c.z) * DX_ + c.w] = i;
  }
}

// Swizzle W[k] (32x32 f32, row-major in->out) into WMMA B-fragment layout:
// fragment (go,h): 32 lanes x 8 dwords; lane l (col N = h*16 + (l&15)),
// dword j holds K = (l<16?0:16)+2j and +1 packed as bf16.
__global__ void prep_weights(const float* __restrict__ W1, const float* __restrict__ W2,
                             const float* __restrict__ W3, const float* __restrict__ W4,
                             unsigned* __restrict__ wfrag) {
  int frag = blockIdx.x;            // 0..175  (88 taps x 2 N-halves)
  int lane = threadIdx.x;           // 0..31
  int go = frag >> 1, h = frag & 1;
  const float* W; int k;
  if (go < 9)       { W = W1; k = go;      }
  else if (go < 34) { W = W2; k = go - 9;  }
  else if (go < 61) { W = W3; k = go - 34; }
  else              { W = W4; k = go - 61; }
  const float* Wk = W + k * 1024;
  int n  = h * 16 + (lane & 15);
  int kb = (lane < 16) ? 0 : 16;
  unsigned* dst = wfrag + frag * 256 + lane * 8;
#pragma unroll
  for (int j = 0; j < 8; ++j) {
    union { unsigned u; __bf16 h2[2]; } pk;
    pk.h2[0] = (__bf16)Wk[(kb + 2*j)     * 32 + n];
    pk.h2[1] = (__bf16)Wk[(kb + 2*j + 1) * 32 + n];
    dst[j] = pk.u;
  }
}

__global__ __launch_bounds__(128) void msconv_kernel(
    const float* __restrict__ feats, const int* __restrict__ idx,
    const int* __restrict__ grid, const unsigned* __restrict__ wfrag,
    const float* __restrict__ gamma, const float* __restrict__ beta,
    float* __restrict__ out)
{
  const int lane = threadIdx.x & 31;
  // wave-uniform group id, forced into an SGPR so group dispatch is scalar
  const int g    = __builtin_amdgcn_readfirstlane((int)(threadIdx.x >> 5));
  const int m    = lane & 15;             // point row within tile
  const int hi   = lane >> 4;             // half-wave selector
  const int base = blockIdx.x * 16;

  const int4 pc = *(const int4*)(idx + (base + m) * 4);   // (b,z,y,x)
  const int lin0 = ((pc.x * DZ_ + pc.y) * DY_ + pc.z) * DX_ + pc.w; // own cell

  int Kg, gob;
  if      (g == 0) { Kg = 9;  gob = 0;  }
  else if (g == 1) { Kg = 25; gob = 9;  }
  else if (g == 2) { Kg = 27; gob = 34; }
  else             { Kg = 27; gob = 61; }

  const int ch0 = g * 32;
  const int c0  = hi * 16;                 // A-fragment chunk base (K offset)

  v8f acc0 = {0.f,0.f,0.f,0.f,0.f,0.f,0.f,0.f};
  v8f acc1 = acc0;

  for (int k = 0; k < Kg; ++k) {
    const int code = OFF88[gob + k];                 // scalar load
    const int oz = (code & 3) - 1;
    const int oy = ((code >> 2) & 7) - 3;
    const int ox = ((code >> 5) & 7) - 3;
    const int dlin = (oz * DY_ + oy) * DX_ + ox;     // scalar

    const int nz = pc.y + oz, ny = pc.z + oy, nx = pc.w + ox;
    int nidx = -1;
    if ((unsigned)nz < (unsigned)DZ_ && (unsigned)ny < (unsigned)DY_ &&
        (unsigned)nx < (unsigned)DX_)
      nidx = grid[lin0 + dlin];

    // Grid occupancy ~1.5%: most taps have no occupied neighbor for ALL 16
    // points of the tile -> A would be all zero. Skip the whole MAC step
    // (wave-uniform branch, EXEC stays full inside for the WMMAs).
    if (__ballot(nidx >= 0) == 0ull) continue;

    // Issue weight-fragment loads first so they overlap the gather below.
    const int go = gob + k;
    const unsigned* w0 = wfrag + (go * 2 + 0) * 256 + lane * 8;
    const unsigned* w1 = wfrag + (go * 2 + 1) * 256 + lane * 8;
    union { unsigned u[8]; v16bf v; } b0, b1;
#pragma unroll
    for (int j = 0; j < 8; ++j) b0.u[j] = w0[j];
#pragma unroll
    for (int j = 0; j < 8; ++j) b1.u[j] = w1[j];

    // Build A fragment (16x32 bf16): lane holds row M=m, K in {c0..c0+15}
    union { unsigned u[8]; __bf16 h[16]; v16bf v; } av;
#pragma unroll
    for (int j = 0; j < 8; ++j) av.u[j] = 0u;
    if (nidx >= 0) {
      const float* fr = feats + (size_t)nidx * 128 + ch0 + c0;
      float4 q0 = *(const float4*)(fr);
      float4 q1 = *(const float4*)(fr + 4);
      float4 q2 = *(const float4*)(fr + 8);
      float4 q3 = *(const float4*)(fr + 12);
      av.h[0]  = (__bf16)q0.x; av.h[1]  = (__bf16)q0.y;
      av.h[2]  = (__bf16)q0.z; av.h[3]  = (__bf16)q0.w;
      av.h[4]  = (__bf16)q1.x; av.h[5]  = (__bf16)q1.y;
      av.h[6]  = (__bf16)q1.z; av.h[7]  = (__bf16)q1.w;
      av.h[8]  = (__bf16)q2.x; av.h[9]  = (__bf16)q2.y;
      av.h[10] = (__bf16)q2.z; av.h[11] = (__bf16)q2.w;
      av.h[12] = (__bf16)q3.x; av.h[13] = (__bf16)q3.y;
      av.h[14] = (__bf16)q3.z; av.h[15] = (__bf16)q3.w;
    }

    acc0 = __builtin_amdgcn_wmma_f32_16x16x32_bf16(false, av.v, false, b0.v,
                                                   (short)0, acc0, false, false);
    acc1 = __builtin_amdgcn_wmma_f32_16x16x32_bf16(false, av.v, false, b1.v,
                                                   (short)0, acc1, false, false);
  }

  // GELU (exact) + residual + GroupNorm + affine + store.
  // C/D layout: acc0[r] = out[point base+r+8*hi][ch0 + m]; acc1 -> channel +16.
  const int   nch = ch0 + m;
  const float gm0 = gamma[nch],      bt0 = beta[nch];
  const float gm1 = gamma[nch + 16], bt1 = beta[nch + 16];
  const float inv_sqrt2 = 0.70710678118654752f;

#pragma unroll
  for (int r = 0; r < 8; ++r) {
    const int    p    = base + r + hi * 8;
    const size_t rowo = (size_t)p * 128 + nch;
    float x0 = acc0[r], x1 = acc1[r];
    x0 = 0.5f * x0 * (1.0f + erff(x0 * inv_sqrt2));
    x1 = 0.5f * x1 * (1.0f + erff(x1 * inv_sqrt2));
    x0 += feats[rowo];
    x1 += feats[rowo + 16];
    float s  = x0 + x1;
    float ss = x0 * x0 + x1 * x1;
#pragma unroll
    for (int mk = 1; mk < 16; mk <<= 1) {   // reduce across the 16-lane half
      s  += __shfl_xor(s,  mk, 16);
      ss += __shfl_xor(ss, mk, 16);
    }
    const float mean = s * (1.0f / 32.0f);
    const float var  = ss * (1.0f / 32.0f) - mean * mean;
    const float inv  = rsqrtf(var + 1e-5f);
    out[rowo]      = (x0 - mean) * inv * gm0 + bt0;
    out[rowo + 16] = (x1 - mean) * inv * gm1 + bt1;
  }
}

extern "C" void kernel_launch(void* const* d_in, const int* in_sizes, int n_in,
                              void* d_out, int out_size, void* d_ws, size_t ws_size,
                              hipStream_t stream) {
  (void)in_sizes; (void)n_in; (void)out_size; (void)ws_size;
  const float* feats = (const float*)d_in[0];
  const int*   idx   = (const int*)d_in[1];
  const float* W1    = (const float*)d_in[2];
  const float* W2    = (const float*)d_in[3];
  const float* W3    = (const float*)d_in[4];
  const float* W4    = (const float*)d_in[5];
  const float* gamma = (const float*)d_in[6];
  const float* beta  = (const float*)d_in[7];
  float* out = (float*)d_out;

  int*      grid  = (int*)d_ws;                                         // 52.48 MB
  unsigned* wfrag = (unsigned*)((char*)d_ws + (size_t)GRID_TOTAL * 4);  // +176 KB

  fill_grid   <<<(GRID_TOTAL + 511) / 512, 512, 0, stream>>>(grid);
  scatter_grid<<<(N_ + 255) / 256,        256, 0, stream>>>(idx, grid);
  prep_weights<<<176, 32, 0, stream>>>(W1, W2, W3, W4, wfrag);
  msconv_kernel<<<N_ / 16, 128, 0, stream>>>(feats, idx, grid, wfrag, gamma, beta, out);
}
